// VectorQuantizer_74629351735513
// MI455X (gfx1250) — matrix operand order repaired and checked by reading, not compiled, not run on hardware
//
#include <hip/hip_runtime.h>
#include <stdint.h>

// ---------------------------------------------------------------------------
// VectorQuantizer forward for MI455X (gfx1250, wave32, WMMA).
// Dominant GEMM (65536x1024, K=256) runs on v_wmma_f32_16x16x32_bf16 with a
// 2-deep software-pipelined B prefetch; codebook / z_q / loss stay fp32.
// One-hot + z_q writes (~340MB) bound overall perf at 23.3 TB/s.
// ---------------------------------------------------------------------------

typedef __attribute__((ext_vector_type(16))) __bf16 v16bf;
typedef __attribute__((ext_vector_type(8)))  float  v8f;

struct alignas(16) U4 { unsigned int x, y, z, w; };
union Frag32B { v16bf v; U4 u[2]; };
struct F2 { float x, y; };

__device__ __forceinline__ unsigned short f2bf(float f) {
  unsigned int u = __float_as_uint(f);
  unsigned int r = u + 0x7FFFu + ((u >> 16) & 1u);   // round-to-nearest-even
  return (unsigned short)(r >> 16);
}

// ---------------------------------------------------------------------------
// K1: codebook[e][c] = dot(map_w[e,:], weights[:,c]) + map_b[e]
//     also emits bf16 copy and cnorm[e] = ||codebook[e]||^2
// ---------------------------------------------------------------------------
__global__ __launch_bounds__(256)
void codebook_kernel(const float* __restrict__ map_w,
                     const float* __restrict__ weights,
                     const float* __restrict__ map_b,
                     float* __restrict__ cb,
                     unsigned short* __restrict__ cbh,
                     float* __restrict__ cnorm)
{
  __shared__ float wrow[256];
  __shared__ float sred[256];
  const int e = blockIdx.x;
  const int c = threadIdx.x;
  float acc = 0.f;
  for (int k0 = 0; k0 < 1024; k0 += 256) {
    wrow[c] = map_w[(size_t)e * 1024 + k0 + c];
    __syncthreads();
#pragma unroll 8
    for (int kk = 0; kk < 256; ++kk)
      acc = fmaf(wrow[kk], weights[(size_t)(k0 + kk) * 256 + c], acc);
    __syncthreads();
  }
  acc += map_b[e];
  cb[(size_t)e * 256 + c]  = acc;
  cbh[(size_t)e * 256 + c] = f2bf(acc);
  sred[c] = acc * acc;
  __syncthreads();
  for (int s = 128; s > 0; s >>= 1) {
    if (c < s) sred[c] += sred[c + s];
    __syncthreads();
  }
  if (c == 0) cnorm[e] = sred[0];
}

// ---------------------------------------------------------------------------
// init: zero histogram + loss accumulator
// ---------------------------------------------------------------------------
__global__ void init_kernel(float* __restrict__ hist, float* __restrict__ lossacc)
{
  const int t = blockIdx.x * blockDim.x + threadIdx.x;
  if (t < 1024) hist[t] = 0.f;
  if (t == 0) *lossacc = 0.f;
}

// ---------------------------------------------------------------------------
// K2: distance GEMM + argmin.
// Block: 32 tokens (one aligned w-run of a (b,h) slab) x all 1024 codes.
// 8 waves x 128-column strips; per wave 2 row-groups x 8 col-tiles of
// v_wmma_f32_16x16x32_bf16, K-loop of 8, flattened into 64 software-
// pipelined steps (B prefetch depth 2, A preloaded one K-tile ahead).
// Epilogue: argmin of cnorm[j] - 2*S via shuffle + LDS reduction.
// ---------------------------------------------------------------------------
__global__ __launch_bounds__(256)
void dist_argmin_kernel(const float* __restrict__ z,
                        const unsigned short* __restrict__ cbh,
                        const float* __restrict__ cnorm,
                        int* __restrict__ idxbuf,
                        float* __restrict__ idx_out_f)
{
  __shared__ unsigned short ldsA[32 * 264];   // [row=w][c], +8 pad vs banks
  __shared__ float redv[32][8];
  __shared__ int   redi[32][8];

  const int n_base = blockIdx.x * 32;         // 2048 blocks
  const int b = n_base >> 10;
  const int h = (n_base & 1023) >> 5;

  // stage A tile (32 x 256) as bf16; coalesced over w for each channel c
  {
    const int t  = threadIdx.x;
    const int w  = t & 31;
    const int c0 = t >> 5;                    // 0..7
    const size_t zbase = (size_t)b * 256 * 1024 + (size_t)h * 32 + w;
#pragma unroll 4
    for (int ci = 0; ci < 32; ++ci) {
      const int c = c0 + ci * 8;
      ldsA[w * 264 + c] = f2bf(z[zbase + (size_t)c * 1024]);
    }
  }
  __syncthreads();

  const int wave = threadIdx.x >> 5;
  const int lane = threadIdx.x & 31;
  const int lr   = lane & 15;
  const int hi   = lane >> 4;
  const int j_base = wave * 128;

  v8f acc[2][8];
  const v8f vzero = {0.f, 0.f, 0.f, 0.f, 0.f, 0.f, 0.f, 0.f};
#pragma unroll
  for (int rg = 0; rg < 2; ++rg)
#pragma unroll
    for (int t = 0; t < 8; ++t) acc[rg][t] = vzero;

  // B fragment: lane = column j (lr), lane-half selects K 0..15 / 16..31
  auto loadB = [&](int step) -> Frag32B {
    const int kt = step >> 3;
    const int t  = step & 7;
    const int j  = j_base + t * 16 + lr;
    const int colB = kt * 32 + hi * 16;
    const U4* pb = reinterpret_cast<const U4*>(cbh + (size_t)j * 256 + colB);
    Frag32B f; f.u[0] = pb[0]; f.u[1] = pb[1];
    return f;
  };
  // A fragment: lanes 0-15 hold K{0..15}, lanes 16-31 hold K{16..31}
  auto loadA = [&](int kt, Frag32B& x0, Frag32B& x1) {
    const int colA = kt * 32 + hi * 16;
    const U4* p0 = reinterpret_cast<const U4*>(&ldsA[lr * 264 + colA]);
    x0.u[0] = p0[0]; x0.u[1] = p0[1];
    const U4* p1 = reinterpret_cast<const U4*>(&ldsA[(16 + lr) * 264 + colA]);
    x1.u[0] = p1[0]; x1.u[1] = p1[1];
  };

  Frag32B a0, a1, a0n, a1n;
  Frag32B bq0, bq1, bq2;
  loadA(0, a0, a1);
  bq0 = loadB(0);
  bq1 = loadB(1);

#pragma unroll
  for (int step = 0; step < 64; ++step) {
    const int kt = step >> 3;
    const int t  = step & 7;
    if (step + 2 < 64) bq2 = loadB(step + 2);          // depth-2 prefetch
    if (t == 6 && kt < 7) loadA(kt + 1, a0n, a1n);     // next K-tile A frags
    acc[0][t] = __builtin_amdgcn_wmma_f32_16x16x32_bf16(
        false, a0.v, false, bq0.v, (short)0, acc[0][t], false, false);
    acc[1][t] = __builtin_amdgcn_wmma_f32_16x16x32_bf16(
        false, a1.v, false, bq0.v, (short)0, acc[1][t], false, false);
    bq0 = bq1;
    if (step + 2 < 64) bq1 = bq2;
    if (t == 7 && kt < 7) { a0 = a0n; a1 = a1n; }
  }

  // per-lane argmin of cnorm[j] - 2*S over this lane's 8 columns per row slot
  float bestv[16];
  int   besti[16];
#pragma unroll
  for (int s = 0; s < 16; ++s) { bestv[s] = 3.4e38f; besti[s] = 0x7fffffff; }

#pragma unroll
  for (int rg = 0; rg < 2; ++rg) {
#pragma unroll
    for (int t = 0; t < 8; ++t) {
      const int j = j_base + t * 16 + lr;
      const float cn = cnorm[j];
#pragma unroll
      for (int r = 0; r < 8; ++r) {
        const float val = cn - 2.0f * acc[rg][t][r];
        const int s = rg * 8 + r;
        if (val < bestv[s] || (val == bestv[s] && j < besti[s])) {
          bestv[s] = val; besti[s] = j;
        }
      }
    }
  }

  // reduce across the 16 lanes of each half (they share rows, differ in column)
#pragma unroll
  for (int s = 0; s < 16; ++s) {
#pragma unroll
    for (int m = 1; m < 16; m <<= 1) {
      const float ov = __shfl_xor(bestv[s], m, 32);
      const int   oi = __shfl_xor(besti[s], m, 32);
      if (ov < bestv[s] || (ov == bestv[s] && oi < besti[s])) {
        bestv[s] = ov; besti[s] = oi;
      }
    }
  }
  if (lr == 0) {
#pragma unroll
    for (int s = 0; s < 16; ++s) {
      const int rg = s >> 3, r = s & 7;
      const int row = rg * 16 + hi * 8 + r;   // C layout: lanes 16-31 hold M+8
      redv[row][wave] = bestv[s];
      redi[row][wave] = besti[s];
    }
  }
  __syncthreads();

  if (threadIdx.x < 32) {
    const int row = threadIdx.x;
    float bv = redv[row][0]; int bi = redi[row][0];
#pragma unroll
    for (int wv = 1; wv < 8; ++wv) {
      const float ov = redv[row][wv]; const int oi = redi[row][wv];
      if (ov < bv || (ov == bv && oi < bi)) { bv = ov; bi = oi; }
    }
    const int n = n_base + row;
    idxbuf[n] = bi;
    idx_out_f[n] = (float)bi;
  }
}

// ---------------------------------------------------------------------------
// K3: z_q gather (fp32 codebook) + NCHW output + loss partial + histogram
// ---------------------------------------------------------------------------
__global__ __launch_bounds__(256)
void gather_kernel(const float* __restrict__ z,
                   const float* __restrict__ cb,
                   const int* __restrict__ idxbuf,
                   float* __restrict__ zq_out,
                   float* __restrict__ hist,
                   float* __restrict__ lossacc)
{
  __shared__ float sred[256];
  const int blk = blockIdx.x;
  const int b = blk >> 5;
  const int h = blk & 31;
  const int t  = threadIdx.x;
  const int w  = t & 31;
  const int c0 = t >> 5;
  const int n = b * 1024 + h * 32 + w;
  const int id = idxbuf[n];
  const size_t zbase = (size_t)b * 256 * 1024 + (size_t)h * 32 + w;
  float lsum = 0.f;
#pragma unroll 4
  for (int ci = 0; ci < 32; ++ci) {
    const int c = c0 + ci * 8;
    const float q = cb[(size_t)id * 256 + c];
    const size_t a = zbase + (size_t)c * 1024;
    const float d = q - z[a];
    lsum += d * d;
    zq_out[a] = q;                            // coalesced over w
  }
  if (c0 == 0) atomicAdd(&hist[id], 1.0f);
  sred[t] = lsum;
  __syncthreads();
  for (int s = 128; s > 0; s >>= 1) {
    if (t < s) sred[t] += sred[t + s];
    __syncthreads();
  }
  if (t == 0) atomicAdd(lossacc, sred[0]);
}

// ---------------------------------------------------------------------------
// K4: one-hot min_encodings [65536 x 1024] (float2: base is 8B-aligned only)
// ---------------------------------------------------------------------------
__global__ __launch_bounds__(256)
void onehot_kernel(const int* __restrict__ idxbuf, float* __restrict__ enc)
{
  const int i = blockIdx.x;
  const int t = threadIdx.x;
  const int id = idxbuf[i];
  const int cbase = t * 4;
  F2 v0 = {0.f, 0.f}, v1 = {0.f, 0.f};
  const int d = id - cbase;
  if (d == 0) v0.x = 1.f;
  else if (d == 1) v0.y = 1.f;
  else if (d == 2) v1.x = 1.f;
  else if (d == 3) v1.y = 1.f;
  F2* p = reinterpret_cast<F2*>(enc + (size_t)i * 1024 + cbase);
  p[0] = v0; p[1] = v1;
}

// ---------------------------------------------------------------------------
// K5: perplexity + loss scalars
// ---------------------------------------------------------------------------
__global__ __launch_bounds__(1024)
void finalize_kernel(const float* __restrict__ hist,
                     const float* __restrict__ lossacc,
                     float* __restrict__ loss_out,
                     float* __restrict__ perp_out)
{
  __shared__ float sred[1024];
  const int t = threadIdx.x;
  const float em = hist[t] * (1.0f / 65536.0f);
  sred[t] = em * logf(em + 1e-10f);
  __syncthreads();
  for (int s = 512; s > 0; s >>= 1) {
    if (t < s) sred[t] += sred[t + s];
    __syncthreads();
  }
  if (t == 0) {
    *perp_out = expf(-sred[0]);
    // loss = (1 + BETA) * mean((z_q - z)^2), BETA = 0.25, mean over 64*256*32*32
    *loss_out = (*lossacc) * (1.25f / 16777216.0f);
  }
}

// ---------------------------------------------------------------------------
extern "C" void kernel_launch(void* const* d_in, const int* in_sizes, int n_in,
                              void* d_out, int out_size, void* d_ws, size_t ws_size,
                              hipStream_t stream)
{
  (void)in_sizes; (void)n_in; (void)out_size; (void)ws_size;
  const float* z       = (const float*)d_in[0];   // [64,256,32,32]
  const float* weights = (const float*)d_in[1];   // [1024,256]
  const float* map_w   = (const float*)d_in[2];   // [1024,1024]
  const float* map_b   = (const float*)d_in[3];   // [1024]
  float* out = (float*)d_out;

  char* ws = (char*)d_ws;
  float*          cb      = (float*)ws;                                   // 1 MB
  unsigned short* cbh     = (unsigned short*)(ws + 1048576);              // 512 KB
  float*          cnorm   = (float*)(ws + 1048576 + 524288);              // 4 KB
  int*            idxbuf  = (int*)(ws + 1048576 + 524288 + 4096);         // 256 KB
  float*          hist    = (float*)(ws + 1048576 + 524288 + 4096 + 262144);
  float*          lossacc = hist + 1024;

  // output tuple layout (flat, return order)
  float* loss_out = out + 0;
  float* zq_out   = out + 1;                      // 16777216 floats
  float* perp_out = out + 16777217;
  float* enc_out  = out + 16777218;               // 67108864 floats
  float* idxf_out = out + 83886082;               // 65536 (indices as float)

  codebook_kernel<<<1024, 256, 0, stream>>>(map_w, weights, map_b, cb, cbh, cnorm);
  init_kernel<<<4, 256, 0, stream>>>(hist, lossacc);
  dist_argmin_kernel<<<2048, 256, 0, stream>>>(z, cbh, cnorm, idxbuf, idxf_out);
  gather_kernel<<<2048, 256, 0, stream>>>(z, cb, idxbuf, zq_out, hist, lossacc);
  onehot_kernel<<<65536, 256, 0, stream>>>(idxbuf, enc_out);
  finalize_kernel<<<1, 1024, 0, stream>>>(hist, lossacc, loss_out, perp_out);
}
